// VAE_8_83330955477078
// MI455X (gfx1250) — compile-verified, matrix operand order checked by reading, not run on hardware
//
#include <hip/hip_runtime.h>
#include <math.h>

#define B   1024
#define S   100
#define H   128
#define SSZ 100
#define IN  2
#define H2  (2*H)      // 256
#define H3  (3*H)      // 384
#define KP  384        // padded K for pfc1 (356 -> 384)

typedef __attribute__((ext_vector_type(16))) _Float16 v16h;
typedef __attribute__((ext_vector_type(8)))  _Float16 v8h;
typedef __attribute__((ext_vector_type(8)))  float    v8f;

// ---------------------------------------------------------------------------
// WMMA fragment load: 16-bit A/B layout for v_wmma_f32_16x16x32_f16.
// Lane L (r=L&15, half=L>>4) holds row r, K = {half*8..half*8+7} U {16+half*8..+7}.
// Requires 16-byte aligned p.
// ---------------------------------------------------------------------------
__device__ __forceinline__ v16h load_frag(const _Float16* __restrict__ p) {
    v8h lo = *(const v8h*)(p);
    v8h hi = *(const v8h*)(p + 16);
    v16h f;
#pragma unroll
    for (int i = 0; i < 8; ++i) { f[i] = lo[i]; f[i + 8] = hi[i]; }
    return f;
}

// ---------------------------------------------------------------------------
// Generic WMMA GEMM: C[M,N] = A[M,K] * W[N,K]^T + bias
// A row-major f16 (lda=K), W row-major f16 (ldw=K). One wave -> 16x64 C tile.
// Requires M%16==0, N%64==0, K%32==0. C (f32) and Ch (f16 mirror) nullable.
// ---------------------------------------------------------------------------
__global__ void k_gemm(const _Float16* __restrict__ A, const _Float16* __restrict__ W,
                       const float* __restrict__ bias, float* __restrict__ C,
                       _Float16* __restrict__ Ch, int M, int N, int K) {
    int gtid = blockIdx.x * blockDim.x + threadIdx.x;
    int wave = gtid >> 5, lane = gtid & 31;
    int ntile = N >> 6;                 // 64-wide N tiles
    int tm = wave / ntile, tn = wave % ntile;
    if (tm * 16 >= M) return;
    int r = lane & 15, half = lane >> 4;

    v8f acc0 = {}, acc1 = {}, acc2 = {}, acc3 = {};
    const _Float16* arow = A + (size_t)(tm * 16 + r) * K + half * 8;
    const _Float16* wbase = W + (size_t)(tn * 64 + r) * K + half * 8;

    for (int k0 = 0; k0 < K; k0 += 32) {
        v16h af = load_frag(arow + k0);
        v16h b0 = load_frag(wbase + k0);
        v16h b1 = load_frag(wbase + (size_t)16 * K + k0);
        v16h b2 = load_frag(wbase + (size_t)32 * K + k0);
        v16h b3 = load_frag(wbase + (size_t)48 * K + k0);
        acc0 = __builtin_amdgcn_wmma_f32_16x16x32_f16(false, af, false, b0, (short)0, acc0, false, false);
        acc1 = __builtin_amdgcn_wmma_f32_16x16x32_f16(false, af, false, b1, (short)0, acc1, false, false);
        acc2 = __builtin_amdgcn_wmma_f32_16x16x32_f16(false, af, false, b2, (short)0, acc2, false, false);
        acc3 = __builtin_amdgcn_wmma_f32_16x16x32_f16(false, af, false, b3, (short)0, acc3, false, false);
    }

    v8f* accs[4] = { &acc0, &acc1, &acc2, &acc3 };
#pragma unroll
    for (int j = 0; j < 4; ++j) {
        int n = tn * 64 + j * 16 + r;
        float bv = bias ? bias[n] : 0.f;
#pragma unroll
        for (int v = 0; v < 8; ++v) {
            int m = tm * 16 + v + 8 * half;     // C layout: VGPR v, half selects M+8
            float val = (*accs[j])[v] + bv;
            size_t o = (size_t)m * N + n;
            if (C)  C[o]  = val;
            if (Ch) Ch[o] = (_Float16)val;
        }
    }
}

// --------------------------- prep / elementwise ----------------------------

__global__ void k_inst_emb(const float* __restrict__ inst, const float* __restrict__ Wi,
                           const float* __restrict__ bi, _Float16* __restrict__ ih16) {
    int idx = blockIdx.x * blockDim.x + threadIdx.x;
    if (idx >= B * S * H) return;
    int bs = idx / H, h = idx % H;
    float v = inst[(size_t)bs * IN] * Wi[h * IN] + inst[(size_t)bs * IN + 1] * Wi[h * IN + 1] + bi[h];
    ih16[idx] = (_Float16)v;
}

__global__ void k_cvt(const float* __restrict__ s, _Float16* __restrict__ d, int n) {
    int i = blockIdx.x * blockDim.x + threadIdx.x;
    if (i < n) d[i] = (_Float16)s[i];
}

// attn_W/ptr_W: [2H,H] stored [k,h] -> two transposed [H,H] f16 halves [h,k]
__global__ void k_wsplitT(const float* __restrict__ W, _Float16* __restrict__ d1,
                          _Float16* __restrict__ d2) {
    int idx = blockIdx.x * blockDim.x + threadIdx.x;
    if (idx >= H * H) return;
    int h = idx / H, k = idx % H;
    d1[idx] = (_Float16)W[(size_t)k * H + h];
    d2[idx] = (_Float16)W[(size_t)(H + k) * H + h];
}

// pfc1_W [2H, 356] -> f16 [2H, 384] zero-padded
__global__ void k_pad_pfc1(const float* __restrict__ src, _Float16* __restrict__ dst) {
    int idx = blockIdx.x * blockDim.x + threadIdx.x;
    if (idx >= H2 * KP) return;
    int n = idx / KP, k = idx % KP;
    dst[idx] = (k < (H2 + SSZ)) ? (_Float16)src[(size_t)n * (H2 + SSZ) + k] : (_Float16)0.f;
}

__global__ void k_zero2(float* __restrict__ a, _Float16* __restrict__ b, int n) {
    int i = blockIdx.x * blockDim.x + threadIdx.x;
    if (i < n) { a[i] = 0.f; b[i] = (_Float16)0.f; }
}

// ref_h = emb_r(instance[b, sol[b,t]]); write f16 ref + into a packed row dst
__global__ void k_emb_ref(const float* __restrict__ inst, const int* __restrict__ sol, int t,
                          const float* __restrict__ W, const float* __restrict__ bias,
                          _Float16* __restrict__ ref16, _Float16* __restrict__ dst16,
                          int ld, int off) {
    int idx = blockIdx.x * blockDim.x + threadIdx.x;
    if (idx >= B * H) return;
    int b = idx / H, h = idx % H;
    int p = sol[(size_t)b * S + t];
    float x0 = inst[((size_t)b * S + p) * IN];
    float x1 = inst[((size_t)b * S + p) * IN + 1];
    _Float16 hv = (_Float16)(x0 * W[h * IN] + x1 * W[h * IN + 1] + bias[h]);
    ref16[idx] = hv;
    dst16[(size_t)b * ld + off + h] = hv;
}

// GRU gate combine (PyTorch order r,z,n); in-place h allowed
__global__ void k_gru(const float* __restrict__ gx, const float* __restrict__ gh,
                      const float* __restrict__ hprev, float* __restrict__ hout,
                      _Float16* __restrict__ hout16) {
    int idx = blockIdx.x * blockDim.x + threadIdx.x;
    if (idx >= B * H) return;
    int b = idx / H, h = idx % H;
    size_t o = (size_t)b * H3 + h;
    float r = 1.f / (1.f + __expf(-(gx[o] + gh[o])));
    float z = 1.f / (1.f + __expf(-(gx[o + H] + gh[o + H])));
    float n = tanhf(gx[o + 2 * H] + r * gh[o + 2 * H]);
    float hv = (1.f - z) * n + z * hprev[idx];
    hout[idx] = hv;
    hout16[idx] = (_Float16)hv;
}

// fused: logits = relu(attnA + q)@v -> softmax(s) -> context = a@inst_h (f16 out)
__global__ void k_attn_ctx(const _Float16* __restrict__ attnA, const float* __restrict__ q,
                           const float* __restrict__ v, const _Float16* __restrict__ ih16,
                           _Float16* __restrict__ dst16, int ld, int off) {
    int b = blockIdx.x, tid = threadIdx.x, lane = tid & 31, wid = tid >> 5;
    __shared__ float sl[S];
    for (int s = wid; s < S; s += 4) {
        float acc = 0.f;
        const _Float16* arow = attnA + ((size_t)b * S + s) * H;
        for (int h = lane; h < H; h += 32)
            acc += fmaxf((float)arow[h] + q[(size_t)b * H + h], 0.f) * v[h];
        for (int o = 16; o; o >>= 1) acc += __shfl_down(acc, o);
        if (lane == 0) sl[s] = acc;
    }
    __syncthreads();
    float mx = -1e30f;
    for (int s = 0; s < S; ++s) mx = fmaxf(mx, sl[s]);
    float sum = 0.f;
    for (int s = 0; s < S; ++s) sum += __expf(sl[s] - mx);
    float inv = 1.f / sum;
    float c = 0.f;
    for (int s = 0; s < S; ++s)
        c += __expf(sl[s] - mx) * inv * (float)ih16[((size_t)b * S + s) * H + tid];
    dst16[(size_t)b * ld + off + tid] = (_Float16)c;
}

// mu/log_var/Z from h2 (one-shot; N=100 so plain f32)
__global__ void k_encfc(const float* __restrict__ h2, const float* __restrict__ W1,
                        const float* __restrict__ b1, const float* __restrict__ W2,
                        const float* __restrict__ b2, const float* __restrict__ eps,
                        float* __restrict__ mu, float* __restrict__ lv,
                        float* __restrict__ zo, float* __restrict__ zbuf) {
    int idx = blockIdx.x * blockDim.x + threadIdx.x;
    if (idx >= B * SSZ) return;
    int b = idx / SSZ, n = idx % SSZ;
    const float* hr = h2 + (size_t)b * H;
    float m = b1[n], l = b2[n];
    for (int k = 0; k < H; ++k) { float hv = hr[k]; m += hv * W1[(size_t)n * H + k]; l += hv * W2[(size_t)n * H + k]; }
    float z = m + eps[idx] * __expf(0.5f * l);
    mu[idx] = m; lv[idx] = l; zo[idx] = z; zbuf[idx] = z;
}

// write Z slice + zero pad into fc_in rows (once; constant across decoder steps)
__global__ void k_fill_Z(const float* __restrict__ zbuf, _Float16* __restrict__ fcin) {
    int idx = blockIdx.x * blockDim.x + threadIdx.x;
    if (idx >= B * H) return;
    int b = idx / H, j = idx % H;
    if (j < SSZ) fcin[(size_t)b * KP + H + j] = (_Float16)zbuf[(size_t)b * SSZ + j];
    else         fcin[(size_t)b * KP + (H2 + SSZ) + (j - SSZ)] = (_Float16)0.f;
}

__global__ void k_mask_init(const int* __restrict__ sol2, float* __restrict__ mask) {
    int idx = blockIdx.x * blockDim.x + threadIdx.x;
    if (idx >= B * S) return;
    int b = idx / S, s = idx % S;
    mask[idx] = (s == sol2[(size_t)b * S]) ? 0.f : 1.f;
}

__global__ void k_ti0(const int* __restrict__ sol2, float* __restrict__ ti) {
    int b = blockIdx.x * blockDim.x + threadIdx.x;
    if (b < B) ti[(size_t)b * S] = (float)sol2[(size_t)b * S];
}

// fused pointer head: tanh(ptrA + p2)@v -> masked softmax -> argmax/logp, update mask
__global__ void k_ptr(const _Float16* __restrict__ ptrA, const float* __restrict__ p2,
                      const float* __restrict__ v, const int* __restrict__ sol2, int t,
                      float* __restrict__ mask, float* __restrict__ ti,
                      float* __restrict__ lp) {
    int b = blockIdx.x, tid = threadIdx.x, lane = tid & 31, wid = tid >> 5;
    __shared__ float sl[S];
    for (int s = wid; s < S; s += 4) {
        float acc = 0.f;
        const _Float16* arow = ptrA + ((size_t)b * S + s) * H;
        for (int h = lane; h < H; h += 32)
            acc += tanhf((float)arow[h] + p2[(size_t)b * H + h]) * v[h];
        for (int o = 16; o; o >>= 1) acc += __shfl_down(acc, o);
        if (lane == 0) sl[s] = acc;
    }
    __syncthreads();
    if (tid == 0) {
        float mx = -1e30f;
        for (int s = 0; s < S; ++s) if (mask[(size_t)b * S + s] > 0.f) mx = fmaxf(mx, sl[s]);
        float sum = 0.f;
        for (int s = 0; s < S; ++s) if (mask[(size_t)b * S + s] > 0.f) sum += __expf(sl[s] - mx);
        int am = 0; float bv = -1e30f;
        for (int s = 0; s < S; ++s) {
            float val = (mask[(size_t)b * S + s] > 0.f) ? sl[s] : -1e30f;
            if (val > bv) { bv = val; am = s; }
        }
        int ptr = sol2[(size_t)b * S + t];
        ti[(size_t)b * S + t] = (float)am;
        lp[(size_t)b * (S - 1) + (t - 1)] = (sl[ptr] - mx) - __logf(sum);
        mask[(size_t)b * S + ptr] = 0.f;
    }
}

// --------------------------------- host ------------------------------------

static inline void gemm(hipStream_t st, const _Float16* A, const _Float16* W,
                        const float* bias, float* C, _Float16* Ch, int M, int N, int K) {
    int waves = (M / 16) * (N / 64);
    k_gemm<<<dim3((waves * 32 + 255) / 256), dim3(256), 0, st>>>(A, W, bias, C, Ch, M, N, K);
}
#define GRID1(n) dim3(((n) + 255) / 256), dim3(256)

extern "C" void kernel_launch(void* const* d_in, const int* in_sizes, int n_in,
                              void* d_out, int out_size, void* d_ws, size_t ws_size,
                              hipStream_t stream) {
    const float* instance = (const float*)d_in[0];
    const int*   sol1     = (const int*)d_in[1];
    const int*   sol2     = (const int*)d_in[2];
    const float* eps      = (const float*)d_in[3];
    const float* emb_i_W  = (const float*)d_in[4];
    const float* emb_i_b  = (const float*)d_in[5];
    const float* emb_r_W  = (const float*)d_in[6];
    const float* emb_r_b  = (const float*)d_in[7];
    const float* attn_W   = (const float*)d_in[8];
    const float* attn_v   = (const float*)d_in[9];
    const float* gru_Wih  = (const float*)d_in[10];
    const float* gru_Whh  = (const float*)d_in[11];
    const float* gru_bih  = (const float*)d_in[12];
    const float* gru_bhh  = (const float*)d_in[13];
    const float* grud_Wih = (const float*)d_in[14];
    const float* grud_Whh = (const float*)d_in[15];
    const float* grud_bih = (const float*)d_in[16];
    const float* grud_bhh = (const float*)d_in[17];
    const float* efc1_W   = (const float*)d_in[18];
    const float* efc1_b   = (const float*)d_in[19];
    const float* efc2_W   = (const float*)d_in[20];
    const float* efc2_b   = (const float*)d_in[21];
    const float* ptr_W    = (const float*)d_in[22];
    const float* ptr_v    = (const float*)d_in[23];
    const float* pfc1_W   = (const float*)d_in[24];
    const float* pfc1_b   = (const float*)d_in[25];
    const float* pfc2_W   = (const float*)d_in[26];
    const float* pfc2_b   = (const float*)d_in[27];
    (void)in_sizes; (void)n_in; (void)ws_size;

    float* out   = (float*)d_out;
    float* o_mu  = out;
    float* o_lv  = out + (size_t)B * SSZ;
    float* o_z   = out + (size_t)2 * B * SSZ;
    float* o_ti  = out + (size_t)3 * B * SSZ;
    float* o_lp  = o_ti + (size_t)B * S;
    (void)out_size;

    char* w = (char*)d_ws; size_t off = 0;
    auto alloc = [&](size_t bytes) -> char* {
        char* p = w + off; off = (off + bytes + 255) & ~(size_t)255; return p;
    };
    _Float16* ih16    = (_Float16*)alloc((size_t)B * S * H * 2);
    _Float16* attnA   = (_Float16*)alloc((size_t)B * S * H * 2);
    _Float16* ptrA    = (_Float16*)alloc((size_t)B * S * H * 2);
    _Float16* gWih16  = (_Float16*)alloc((size_t)H3 * H * 2);
    _Float16* gWhh16  = (_Float16*)alloc((size_t)H3 * H * 2);
    _Float16* dWih16  = (_Float16*)alloc((size_t)H3 * H2 * 2);
    _Float16* dWhh16  = (_Float16*)alloc((size_t)H3 * H * 2);
    _Float16* aW1t    = (_Float16*)alloc((size_t)H * H * 2);
    _Float16* aW2t    = (_Float16*)alloc((size_t)H * H * 2);
    _Float16* pW1t    = (_Float16*)alloc((size_t)H * H * 2);
    _Float16* pW2t    = (_Float16*)alloc((size_t)H * H * 2);
    _Float16* pfc1h   = (_Float16*)alloc((size_t)H2 * KP * 2);
    _Float16* pfc2h   = (_Float16*)alloc((size_t)H * H2 * 2);
    float*    h1      = (float*)alloc((size_t)B * H * 4);
    float*    h2      = (float*)alloc((size_t)B * H * 4);
    float*    gx      = (float*)alloc((size_t)B * H3 * 4);
    float*    gh      = (float*)alloc((size_t)B * H3 * 4);
    float*    q       = (float*)alloc((size_t)B * H * 4);
    float*    p2      = (float*)alloc((size_t)B * H * 4);
    float*    zbuf    = (float*)alloc((size_t)B * SSZ * 4);
    float*    mask    = (float*)alloc((size_t)B * S * 4);
    _Float16* h1h     = (_Float16*)alloc((size_t)B * H * 2);
    _Float16* h2h     = (_Float16*)alloc((size_t)B * H * 2);
    _Float16* ref16   = (_Float16*)alloc((size_t)B * H * 2);
    _Float16* rnn16   = (_Float16*)alloc((size_t)B * H2 * 2);
    _Float16* fcin16  = (_Float16*)alloc((size_t)B * KP * 2);
    _Float16* fc1h    = (_Float16*)alloc((size_t)B * H2 * 2);
    _Float16* fc2h    = (_Float16*)alloc((size_t)B * H * 2);

    // ---- one-time prep ----
    k_inst_emb<<<GRID1(B * S * H), 0, stream>>>(instance, emb_i_W, emb_i_b, ih16);
    k_cvt<<<GRID1(H3 * H), 0, stream>>>(gru_Wih, gWih16, H3 * H);
    k_cvt<<<GRID1(H3 * H), 0, stream>>>(gru_Whh, gWhh16, H3 * H);
    k_cvt<<<GRID1(H3 * H2), 0, stream>>>(grud_Wih, dWih16, H3 * H2);
    k_cvt<<<GRID1(H3 * H), 0, stream>>>(grud_Whh, dWhh16, H3 * H);
    k_cvt<<<GRID1(H * H2), 0, stream>>>(pfc2_W, pfc2h, H * H2);
    k_wsplitT<<<GRID1(H * H), 0, stream>>>(attn_W, aW1t, aW2t);
    k_wsplitT<<<GRID1(H * H), 0, stream>>>(ptr_W, pW1t, pW2t);
    k_pad_pfc1<<<GRID1(H2 * KP), 0, stream>>>(pfc1_W, pfc1h);
    // hoisted scan-invariant GEMMs: inst_h @ attn_W[:H], inst_h @ ptr_W[:H]
    gemm(stream, ih16, aW1t, nullptr, nullptr, attnA, B * S, H, H);
    gemm(stream, ih16, pW1t, nullptr, nullptr, ptrA, B * S, H, H);
    k_zero2<<<GRID1(B * H), 0, stream>>>(h1, h1h, B * H);
    k_zero2<<<GRID1(B * H), 0, stream>>>(h2, h2h, B * H);
    k_emb_ref<<<GRID1(B * H), 0, stream>>>(instance, sol1, 0, emb_r_W, emb_r_b,
                                           ref16, rnn16, H2, 0);

    // ---- encoder scan ----
    for (int t = 1; t < S; ++t) {
        gemm(stream, ref16, gWih16, gru_bih, gx, nullptr, B, H3, H);
        gemm(stream, h1h, gWhh16, gru_bhh, gh, nullptr, B, H3, H);
        k_gru<<<GRID1(B * H), 0, stream>>>(gx, gh, h1, h1, h1h);
        gemm(stream, h1h, aW2t, nullptr, q, nullptr, B, H, H);
        k_attn_ctx<<<dim3(B), dim3(H), 0, stream>>>(attnA, q, attn_v, ih16, rnn16, H2, H);
        k_emb_ref<<<GRID1(B * H), 0, stream>>>(instance, sol1, t, emb_r_W, emb_r_b,
                                               ref16, rnn16, H2, 0);
        gemm(stream, rnn16, dWih16, grud_bih, gx, nullptr, B, H3, H2);
        gemm(stream, h2h, dWhh16, grud_bhh, gh, nullptr, B, H3, H);
        k_gru<<<GRID1(B * H), 0, stream>>>(gx, gh, h2, h2, h2h);
    }

    // ---- latent + decoder init ----
    k_encfc<<<GRID1(B * SSZ), 0, stream>>>(h2, efc1_W, efc1_b, efc2_W, efc2_b, eps,
                                           o_mu, o_lv, o_z, zbuf);
    k_fill_Z<<<GRID1(B * H), 0, stream>>>(zbuf, fcin16);
    k_zero2<<<GRID1(B * H), 0, stream>>>(h1, h1h, B * H);
    k_mask_init<<<GRID1(B * S), 0, stream>>>(sol2, mask);
    k_ti0<<<GRID1(B), 0, stream>>>(sol2, o_ti);

    // ---- decoder scan (teacher forcing) ----
    for (int t = 1; t < S; ++t) {
        k_emb_ref<<<GRID1(B * H), 0, stream>>>(instance, sol2, t - 1, emb_r_W, emb_r_b,
                                               ref16, fcin16, KP, H2 + SSZ - H); // off=228
        gemm(stream, ref16, gWih16, gru_bih, gx, nullptr, B, H3, H);
        gemm(stream, h1h, gWhh16, gru_bhh, gh, nullptr, B, H3, H);
        k_gru<<<GRID1(B * H), 0, stream>>>(gx, gh, h1, h1, h1h);
        gemm(stream, h1h, aW2t, nullptr, q, nullptr, B, H, H);
        k_attn_ctx<<<dim3(B), dim3(H), 0, stream>>>(attnA, q, attn_v, ih16, fcin16, KP, 0);
        gemm(stream, fcin16, pfc1h, pfc1_b, nullptr, fc1h, B, H2, KP);
        gemm(stream, fc1h, pfc2h, pfc2_b, nullptr, fc2h, B, H, H2);
        gemm(stream, fc2h, pW2t, nullptr, p2, nullptr, B, H, H);
        k_ptr<<<dim3(B), dim3(H), 0, stream>>>(ptrA, p2, ptr_v, sol2, t, mask, o_ti, o_lp);
    }
}
